// GatedLinearAttentionARMA_61040075210838
// MI455X (gfx1250) — matrix operand order repaired and checked by reading, not compiled, not run on hardware
//
#include <hip/hip_runtime.h>
#include <hip/hip_bf16.h>
#include <math.h>

// ---------------------------------------------------------------------------
// GatedLinearAttentionARMA for MI455X (gfx1250, wave32, WMMA).
// B=2, L=1024, D=1024, H=16, d=64.
//
//  Stage 1: Q = X q1^T, K = X k1^T, K2s = sigmoid((X k2^T)*0.02/sqrt(D))
//           via bf16 WMMA GEMM with fp32 accumulation.
//  Stage 2: per-(b,h) sequential gated scan + ARMA correction (VALU).
//  Stage 3: out = Y c_proj^T via the same WMMA GEMM.
// ---------------------------------------------------------------------------

typedef __attribute__((ext_vector_type(16))) __bf16 v16bf;
typedef __attribute__((ext_vector_type(8)))  float  v8f;

#define BB 2
#define LL 1024
#define DD 1024
#define HH 16
#define dh 64
#define MM (BB * LL)          // 2048 rows in all GEMMs

__device__ inline float fast_rcp(float x) { return __builtin_amdgcn_rcpf(x); }
__device__ inline float fast_sigmoid(float x) {
  return fast_rcp(1.0f + __expf(-x));           // v_exp_f32 + v_rcp_f32
}

// ---------------------------------------------------------------------------
// Fragment loaders (fp32 memory -> bf16 WMMA operand registers).
//
// A (16x32, MxK) per ISA 7.12.2: lane = half*16 + m ; v16bf elements
//   e in [0,8)  -> K = k0 + half*8 + e
//   e in [8,16) -> K = k0 + 16 + half*8 + (e-8)
// ---------------------------------------------------------------------------
__device__ inline v16bf load_frag_a(const float* __restrict__ rows, int ld,
                                    int lane, int k0) {
  const int m    = lane & 15;
  const int half = lane >> 4;
  const float* p = rows + (size_t)m * ld + k0 + half * 8;
  v16bf f;
#pragma unroll
  for (int e = 0; e < 8; ++e) f[e] = (__bf16)p[e];
#pragma unroll
  for (int e = 0; e < 8; ++e) f[8 + e] = (__bf16)p[16 + e];
  return f;
}

// B (32x16, KxN). We compute X * W^T, so B[k][n] = W[n][k]: lane's column n
// is weight row n. Same K order as A: e in [0,8) -> K = k0 + half*8 + e,
// e in [8,16) -> K = k0 + 16 + half*8 + (e-8).
__device__ inline v16bf load_frag_b(const float* __restrict__ wrows, int ld,
                                    int lane, int k0) {
  const int n    = lane & 15;
  const int half = lane >> 4;
  const float* p = wrows + (size_t)n * ld + k0 + half * 8;
  v16bf f;
#pragma unroll
  for (int e = 0; e < 8; ++e) f[e] = (__bf16)p[e];
#pragma unroll
  for (int e = 0; e < 8; ++e) f[8 + e] = (__bf16)p[16 + e];
  return f;
}

// ---------------------------------------------------------------------------
// C[M,N] = A[M,K] * W[N,K]^T.  One wave computes a 32x64 tile:
// 2 A-fragments x 4 B-fragments -> 8 v_wmma_f32_16x16x32_bf16 per K-step.
// mode 1: epilogue sigmoid(x * 0.02/sqrt(D))  (for K2).
// ---------------------------------------------------------------------------
__global__ __launch_bounds__(128)
void wmma_gemm_xwt(const float* __restrict__ A, const float* __restrict__ W,
                   float* __restrict__ C, int M, int N, int Kd, int mode) {
  const int lane = threadIdx.x & 31;
  const int wave = threadIdx.x >> 5;
  const int tilesN = N >> 6;                    // 64-wide tiles
  const int tile   = blockIdx.x * 4 + wave;
  const int numTiles = (M >> 5) * tilesN;
  if (tile >= numTiles) return;                 // wave-uniform
  const int m0 = (tile / tilesN) << 5;
  const int n0 = (tile % tilesN) << 6;

  v8f acc[2][4];
#pragma unroll
  for (int fm = 0; fm < 2; ++fm)
#pragma unroll
    for (int fn = 0; fn < 4; ++fn)
#pragma unroll
      for (int r = 0; r < 8; ++r) acc[fm][fn][r] = 0.0f;

  for (int k0 = 0; k0 < Kd; k0 += 32) {
    // Prefetch next K-step of this wave's A rows (global_prefetch_b8).
    if (k0 + 32 < Kd) {
      __builtin_prefetch(A + (size_t)(m0 + (lane & 15)) * Kd + k0 + 32, 0, 3);
      __builtin_prefetch(A + (size_t)(m0 + 16 + (lane & 15)) * Kd + k0 + 32, 0, 3);
    }
    v16bf a0 = load_frag_a(A + (size_t)m0 * Kd, Kd, lane, k0);
    v16bf a1 = load_frag_a(A + (size_t)(m0 + 16) * Kd, Kd, lane, k0);
    v16bf b0 = load_frag_b(W + (size_t)(n0 +  0) * Kd, Kd, lane, k0);
    v16bf b1 = load_frag_b(W + (size_t)(n0 + 16) * Kd, Kd, lane, k0);
    v16bf b2 = load_frag_b(W + (size_t)(n0 + 32) * Kd, Kd, lane, k0);
    v16bf b3 = load_frag_b(W + (size_t)(n0 + 48) * Kd, Kd, lane, k0);

    acc[0][0] = __builtin_amdgcn_wmma_f32_16x16x32_bf16(false, a0, false, b0,
                (short)0, acc[0][0], false, false);
    acc[0][1] = __builtin_amdgcn_wmma_f32_16x16x32_bf16(false, a0, false, b1,
                (short)0, acc[0][1], false, false);
    acc[0][2] = __builtin_amdgcn_wmma_f32_16x16x32_bf16(false, a0, false, b2,
                (short)0, acc[0][2], false, false);
    acc[0][3] = __builtin_amdgcn_wmma_f32_16x16x32_bf16(false, a0, false, b3,
                (short)0, acc[0][3], false, false);
    acc[1][0] = __builtin_amdgcn_wmma_f32_16x16x32_bf16(false, a1, false, b0,
                (short)0, acc[1][0], false, false);
    acc[1][1] = __builtin_amdgcn_wmma_f32_16x16x32_bf16(false, a1, false, b1,
                (short)0, acc[1][1], false, false);
    acc[1][2] = __builtin_amdgcn_wmma_f32_16x16x32_bf16(false, a1, false, b2,
                (short)0, acc[1][2], false, false);
    acc[1][3] = __builtin_amdgcn_wmma_f32_16x16x32_bf16(false, a1, false, b3,
                (short)0, acc[1][3], false, false);
  }

  // C/D layout: lane half -> M offset 0/8, VGPR r -> +M, lane&15 -> N
  const float kSig = 0.02f / 32.0f;             // 0.02 / sqrt(1024)
#pragma unroll
  for (int fm = 0; fm < 2; ++fm) {
#pragma unroll
    for (int fn = 0; fn < 4; ++fn) {
      const int n     = n0 + fn * 16 + (lane & 15);
      const int mbase = m0 + fm * 16 + (lane >> 4) * 8;
#pragma unroll
      for (int r = 0; r < 8; ++r) {
        float v = acc[fm][fn][r];
        if (mode == 1) v = fast_sigmoid(v * kSig);
        C[(size_t)(mbase + r) * N + n] = v;
      }
    }
  }
}

// ---------------------------------------------------------------------------
// Gated scan + ARMA correction. One block per (b,h); 64 threads; thread e
// owns state columns S1[0..63][e], S2[0..63][e] in registers.
// Dot products use 4 independent partial accumulators to keep the dependent
// FMA chain on the recurrent critical path at depth 16, not 64.
// ---------------------------------------------------------------------------
__global__ __launch_bounds__(64)
void gla_arma_scan(const float* __restrict__ Q, const float* __restrict__ K,
                   const float* __restrict__ K2, const float* __restrict__ X,
                   const float* __restrict__ gw, const float* __restrict__ sw,
                   float* __restrict__ Y) {
  const int bh = blockIdx.x;
  const int b  = bh / HH;
  const int h  = bh % HH;
  const int t  = threadIdx.x;                   // column index e

  __shared__ float sQ[dh], sK[dh], sV[dh], sK2[dh];
  __shared__ float sQm[dh], sK2p[dh];           // previous-step Qm / K2
  __shared__ float sGW[dh], sSW[dh];

  sGW[t] = gw[t];
  sSW[t] = sw[t];

  float S1[dh], S2[dh];
#pragma unroll
  for (int i = 0; i < dh; ++i) { S1[i] = 0.0f; S2[i] = 0.0f; }
  float la = 0.0f;                              // cumsum(log G)
  float O1prev = 0.0f;

  const size_t base = (size_t)b * LL * DD + (size_t)h * dh;

  for (int l = 0; l < LL; ++l) {
    const size_t off = base + (size_t)l * DD;
    __syncthreads();                            // prev-step readers done
    sQ[t]  = Q[off + t];
    sK[t]  = K[off + t];
    sV[t]  = X[off + t];                        // V = X head slice
    sK2[t] = K2[off + t];
    __syncthreads();

    // Per-(b,l,h) scalars (computed redundantly in all lanes).
    float gd[4] = {0.f, 0.f, 0.f, 0.f};
    float rd[4] = {0.f, 0.f, 0.f, 0.f};
#pragma unroll
    for (int i = 0; i < dh; i += 4) {
#pragma unroll
      for (int j = 0; j < 4; ++j) {
        gd[j] = fmaf(sV[i + j], sGW[i + j], gd[j]);
        rd[j] = fmaf(sK[i + j], sSW[i + j], rd[j]);
      }
    }
    const float gdot = (gd[0] + gd[1]) + (gd[2] + gd[3]);
    const float rdot = (rd[0] + rd[1]) + (rd[2] + rd[3]);

    float g = fmaxf(fast_sigmoid(gdot), 1e-6f);
    la += __logf(g);
    const float lac = fminf(fmaxf(la, -30.0f), 30.0f);
    const float Gc  = __expf(lac) + 1e-6f;
    const float R   = rdot * fast_sigmoid(rdot);       // silu
    const float cs  = R * fast_rcp(Gc);

    // S1 += K ⊗ V * (R/Gc)  (inclusive), O1 = Gc * (Q · S1[:,e])
    const float ve = sV[t];
    const float vc = ve * cs;
    float a1[4] = {0.f, 0.f, 0.f, 0.f};
#pragma unroll
    for (int i = 0; i < dh; i += 4) {
#pragma unroll
      for (int j = 0; j < 4; ++j) {
        S1[i + j] = fmaf(sK[i + j], vc, S1[i + j]);
        a1[j]     = fmaf(sQ[i + j], S1[i + j], a1[j]);
      }
    }
    const float O1 = Gc * ((a1[0] + a1[1]) + (a1[2] + a1[3]));

    // ARMA: E[l-1] = V[l] - O1[l-1]; S2 += K2[l-1] ⊗ E[l-1]; O2 = Qm[l-1]·S2
    float O2 = 0.0f;
    if (l > 0) {
      const float Ep = ve - O1prev;
      float a2[4] = {0.f, 0.f, 0.f, 0.f};
#pragma unroll
      for (int i = 0; i < dh; i += 4) {
#pragma unroll
        for (int j = 0; j < 4; ++j) {
          S2[i + j] = fmaf(sK2p[i + j], Ep, S2[i + j]);
          a2[j]     = fmaf(sQm[i + j], S2[i + j], a2[j]);
        }
      }
      O2 = (a2[0] + a2[1]) + (a2[2] + a2[3]);
    }

    Y[off + t] = O1 + O2;
    O1prev = O1;

    __syncthreads();                            // O2 readers done
    // Qm = -leaky_relu(Q * (-1/sqrt(d)), 0.02)
    const float qv = sQ[t] * (-0.125f);
    sQm[t]  = -((qv >= 0.0f) ? qv : 0.02f * qv);
    sK2p[t] = sK2[t];
  }
}

// ---------------------------------------------------------------------------
extern "C" void kernel_launch(void* const* d_in, const int* in_sizes, int n_in,
                              void* d_out, int out_size, void* d_ws,
                              size_t ws_size, hipStream_t stream) {
  const float* X   = (const float*)d_in[0];
  const float* q1w = (const float*)d_in[1];
  const float* k1w = (const float*)d_in[2];
  const float* k2w = (const float*)d_in[3];
  const float* gww = (const float*)d_in[4];
  const float* sww = (const float*)d_in[5];
  const float* cpw = (const float*)d_in[6];
  float* out = (float*)d_out;

  float* ws  = (float*)d_ws;
  const size_t mat = (size_t)MM * DD;           // 2M floats
  float* Qb  = ws;
  float* Kb  = ws + mat;
  float* K2b = ws + 2 * mat;
  float* Yb  = ws + 3 * mat;

  const int numTiles = (MM >> 5) * (DD >> 6);   // 64 * 16 = 1024
  dim3 gblk(128);
  dim3 ggrd((numTiles + 3) / 4);                // 4 waves (tiles) per block

  wmma_gemm_xwt<<<ggrd, gblk, 0, stream>>>(X, q1w, Qb,  MM, DD, DD, 0);
  wmma_gemm_xwt<<<ggrd, gblk, 0, stream>>>(X, k1w, Kb,  MM, DD, DD, 0);
  wmma_gemm_xwt<<<ggrd, gblk, 0, stream>>>(X, k2w, K2b, MM, DD, DD, 1);

  gla_arma_scan<<<dim3(BB * HH), dim3(dh), 0, stream>>>(Qb, Kb, K2b, X,
                                                        gww, sww, Yb);

  wmma_gemm_xwt<<<ggrd, gblk, 0, stream>>>(Yb, cpw, out, MM, DD, DD, 0);
}
